// FusedDenseLayer_69879117906309
// MI455X (gfx1250) — matrix operand order repaired and checked by reading, not compiled, not run on hardware
//
#include <hip/hip_runtime.h>
#include <hip/hip_bf16.h>

typedef __attribute__((ext_vector_type(16))) __bf16 v16bf;
typedef __attribute__((ext_vector_type(8)))  float  v8f;

#define IN_C   256
#define GROW   32
#define HWDIM  56
#define PLANE  (HWDIM*HWDIM)        // 3136
#define CTOT   288
#define CSTR   264                  // 256 + 8 pad (bf16 elems) to spread LDS banks
#define HALO   100                  // 10x10 halo positions

__device__ __forceinline__ unsigned short f32_to_bf16(float f) {
    unsigned int u = __builtin_bit_cast(unsigned int, f);
    u += 0x7FFFu + ((u >> 16) & 1u);       // round-to-nearest-even
    return (unsigned short)(u >> 16);
}

// ---------------------------------------------------------------------------
// Kernel 1: pack conv weights (OIHW fp32) -> bf16 WMMA chunk layout in d_ws.
// wpack[(q*32 + m)*32 + kk], q = tap*8 + cgrp, k = q*32+kk, c = cgrp*32+kk.
// ---------------------------------------------------------------------------
__global__ void pack_weights_kernel(const float* __restrict__ w,
                                    unsigned short* __restrict__ wp) {
    int idx = blockIdx.x * 256 + threadIdx.x;          // 0 .. 73727
    if (idx >= GROW * IN_C * 9) return;
    int kk = idx & 31;
    int m  = (idx >> 5) & 31;
    int q  = idx >> 10;                                // 0..71
    int cg = q & 7;
    int t  = q >> 3;                                   // tap 0..8
    int r  = t / 3, s = t - 3 * r;
    int c  = cg * 32 + kk;
    float v = w[((m * IN_C + c) * 3 + r) * 3 + s];
    wp[idx] = f32_to_bf16(v);
}

// ---------------------------------------------------------------------------
// Kernel 2: copy the untouched 256-channel block into d_out (float4).
// ---------------------------------------------------------------------------
__global__ void copy_base_kernel(const float4* __restrict__ in,
                                 float4* __restrict__ out) {
    long long id = (long long)blockIdx.x * 256 + threadIdx.x;   // over 32*256*784
    const long long PER_N_IN = (long long)IN_C * (PLANE / 4);   // 200704
    const long long PER_N_TOT = (long long)CTOT * (PLANE / 4);  // 225792
    if (id >= 32 * PER_N_IN) return;
    long long n   = id / PER_N_IN;
    long long rem = id - n * PER_N_IN;
    long long off = n * PER_N_TOT + rem;
    out[off] = in[off];
}

// ---------------------------------------------------------------------------
// Kernel 3: fused BN + ReLU + 3x3 conv via implicit GEMM with bf16 WMMA.
// Grid (7,7,32); 128 threads = 4 waves. Tile: 32 out-ch x 8x8 spatial.
// ---------------------------------------------------------------------------
__global__ __launch_bounds__(128) void fused_bn_relu_conv_kernel(
    const float* __restrict__ x,
    const float* __restrict__ bnw, const float* __restrict__ bnb,
    const float* __restrict__ rmean, const float* __restrict__ rvar,
    const unsigned short* __restrict__ wp,
    float* __restrict__ out) {

    __shared__ unsigned short y_lds[HALO * CSTR];   // [pos][c] bf16, 52.8 KB
    __shared__ float s_scale[IN_C];
    __shared__ float s_shift[IN_C];

    const int tid  = threadIdx.x;
    const int lane = tid & 31;
    const int wid  = tid >> 5;
    const int tx = blockIdx.x, ty = blockIdx.y, n = blockIdx.z;

    // --- BN folded scale/shift ---
    for (int c = tid; c < IN_C; c += 128) {
        float sc = bnw[c] * rsqrtf(rvar[c] + 1e-5f);
        s_scale[c] = sc;
        s_shift[c] = bnb[c] - rmean[c] * sc;
    }
    __syncthreads();

    // --- stage BN+ReLU'd 10x10 halo, all 256 channels, as bf16 [pos][c] ---
    const int gy0 = ty * 8 - 1, gx0 = tx * 8 - 1;
    const float* xin = x + (long long)n * CTOT * PLANE;
    for (int e = tid; e < IN_C * HALO; e += 128) {      // exactly 200 iters/thread
        int c  = e / HALO;
        int h  = e - c * HALO;
        int hy = h / 10;
        int hx = h - hy * 10;
        int gy = gy0 + hy, gx = gx0 + hx;
        unsigned short val = 0;                         // SAME zero padding
        if ((unsigned)gy < (unsigned)HWDIM && (unsigned)gx < (unsigned)HWDIM) {
            float v = xin[c * PLANE + gy * HWDIM + gx];
            v = fmaxf(v * s_scale[c] + s_shift[c], 0.0f);
            val = f32_to_bf16(v);
        }
        y_lds[h * CSTR + c] = val;
    }
    __syncthreads();

    // --- per-wave WMMA: 16 spatial cols (2 rows of 8), 32 out channels ---
    v8f acc0 = {};     // m = 0..15
    v8f acc1 = {};     // m = 16..31
    const int p   = lane & 15;                 // spatial column of B/D
    const int py  = (wid << 1) + (p >> 3);     // tile-local y (0..7)
    const int px  = p & 7;                     // tile-local x
    const int kb  = lane >> 4;                 // lane-half selector
    const int mrow = lane & 15;                // A-fragment row within 16

    union Frag { v16bf v; uint4 u4[2]; };

    for (int r = 0; r < 3; ++r) {
        for (int s = 0; s < 3; ++s) {
            const int t = r * 3 + s;
            const int hb = ((py + r) * 10 + (px + s)) * CSTR;   // halo pos base
            #pragma unroll
            for (int cg = 0; cg < 8; ++cg) {
                const int c0 = cg * 32;
                // B fragment: lanes<16: K{0..7,16..23}; lanes>=16: K{8..15,24..31}
                Frag b;
                const uint4* bp =
                    reinterpret_cast<const uint4*>(&y_lds[hb + c0 + kb * 8]);
                b.u4[0] = bp[0];
                b.u4[1] = bp[2];

                // A fragments: lanes<16: K{0..7,16..23}; lanes>=16: K{8..15,24..31}
                const int q = t * 8 + cg;
                const unsigned short* a0p = wp + ((q * 32 + mrow) * 32 + kb * 8);
                const unsigned short* a1p = a0p + 16 * 32;       // rows m+16
                Frag a0, a1;
                a0.u4[0] = *reinterpret_cast<const uint4*>(a0p);
                a0.u4[1] = *reinterpret_cast<const uint4*>(a0p + 16);
                a1.u4[0] = *reinterpret_cast<const uint4*>(a1p);
                a1.u4[1] = *reinterpret_cast<const uint4*>(a1p + 16);

                acc0 = __builtin_amdgcn_wmma_f32_16x16x32_bf16(
                    false, a0.v, false, b.v, (short)0, acc0, false, false);
                acc1 = __builtin_amdgcn_wmma_f32_16x16x32_bf16(
                    false, a1.v, false, b.v, (short)0, acc1, false, false);
            }
        }
    }

    // --- scatter D tiles: out[n][256+m][oy][ox], documented C/D layout ---
    const int oy = ty * 8 + py, ox = tx * 8 + px;
    float* ob = out + (long long)n * CTOT * PLANE + (long long)IN_C * PLANE
                    + oy * HWDIM + ox;
    const int mof = kb * 8;
    #pragma unroll
    for (int rr = 0; rr < 8; ++rr) {
        ob[(rr + mof) * PLANE]      = acc0[rr];
        ob[(rr + mof + 16) * PLANE] = acc1[rr];
    }
}

// ---------------------------------------------------------------------------
extern "C" void kernel_launch(void* const* d_in, const int* in_sizes, int n_in,
                              void* d_out, int out_size, void* d_ws, size_t ws_size,
                              hipStream_t stream) {
    const float* all_features = (const float*)d_in[0];
    const float* bn_weight    = (const float*)d_in[1];
    const float* bn_bias      = (const float*)d_in[2];
    const float* running_mean = (const float*)d_in[3];
    const float* running_var  = (const float*)d_in[4];
    const float* conv_weight  = (const float*)d_in[5];
    float* out = (float*)d_out;
    unsigned short* wpack = (unsigned short*)d_ws;   // 73728 bf16 = 144 KB

    // 1) pack weights to bf16 fragment layout (73728 elements)
    pack_weights_kernel<<<288, 256, 0, stream>>>(conv_weight, wpack);

    // 2) copy first 256 channels through (float4 granularity)
    {
        long long total4 = 32LL * IN_C * (PLANE / 4);     // 6,422,528
        int blocks = (int)((total4 + 255) / 256);
        copy_base_kernel<<<blocks, 256, 0, stream>>>(
            (const float4*)all_features, (float4*)out);
    }

    // 3) fused BN+ReLU+conv, WMMA
    fused_bn_relu_conv_kernel<<<dim3(7, 7, 32), 128, 0, stream>>>(
        all_features, bn_weight, bn_bias, running_mean, running_var,
        wpack, out);
}
